// Network3layers_82987358093486
// MI455X (gfx1250) — matrix operand order repaired and checked by reading, not compiled, not run on hardware
//
#include <hip/hip_runtime.h>
#include <math.h>

typedef __attribute__((ext_vector_type(16))) _Float16 v16h;
typedef __attribute__((ext_vector_type(8)))  float    v8f;

#define WAVE 32

// ---------------------------------------------------------------------------
// K0: Wall[r,i,o] = sum_k relu(w1[r,k]) * w2[k, i*32+o]    (per original node r)
// ---------------------------------------------------------------------------
__global__ void k_precompute_wall(const float* __restrict__ w1,
                                  const float* __restrict__ w2,
                                  float* __restrict__ wall, int Ci) {
  int r = blockIdx.x;
  float hk[8];
  for (int k = 0; k < 8; ++k) { float v = w1[r * 8 + k]; hk[k] = v > 0.f ? v : 0.f; }
  int tot = Ci * 32;
  for (int t = threadIdx.x; t < tot; t += blockDim.x) {
    float acc = 0.f;
    for (int k = 0; k < 8; ++k) acc += hk[k] * w2[k * tot + t];
    wall[(size_t)r * tot + t] = acc;
  }
}

// ---------------------------------------------------------------------------
// K1: xt[b,n,o] = x[b,n,:] . Wall[orig[b,n]][:,o], written as f16 (K-padded)
// one wave per padded node row; lane = output channel (Co == 32 == wave32)
// ---------------------------------------------------------------------------
__global__ void k_conv_xt(const float* __restrict__ xin,    // [B,N,Ci]
                          const int*   __restrict__ orig,   // [B,N] or null (identity)
                          const float* __restrict__ wall,   // [116,Ci,32]
                          _Float16*    __restrict__ xtH,    // [B,Npad,32]
                          int N, int Npad, int Ci) {
  int w    = blockIdx.x * (blockDim.x / WAVE) + (threadIdx.x / WAVE);
  int lane = threadIdx.x & (WAVE - 1);
  int b = w / Npad, n = w % Npad;
  _Float16* dst = xtH + ((size_t)b * Npad + n) * 32 + lane;
  if (n >= N) { *dst = (_Float16)0.f; return; }
  int r = orig ? orig[b * N + n] : n;
  const float* xrow = xin + ((size_t)b * N + n) * Ci;
  const float* wl   = wall + (size_t)r * Ci * 32 + lane;
  float acc = 0.f;
  for (int i = 0; i < Ci; ++i) acc += xrow[i] * wl[i * 32];
  *dst = (_Float16)acc;
}

// ---------------------------------------------------------------------------
// K2a: self-loop fill + masked row softmax + degree. One wave per padded row.
// alpha written f16 zero-padded to [rowsPad, Npad] for WMMA consumption.
// ---------------------------------------------------------------------------
__global__ void k_softmax(const float* __restrict__ adj,    // [B,N,N]
                          _Float16*    __restrict__ alphaH, // [B,rowsPad,Npad]
                          float*       __restrict__ deg,    // [B,rowsPad]
                          int N, int Npad, int rowsPad) {
  int w    = blockIdx.x * (blockDim.x / WAVE) + (threadIdx.x / WAVE);
  int lane = threadIdx.x & (WAVE - 1);
  int b = w / rowsPad, i = w % rowsPad;
  _Float16* arow = alphaH + ((size_t)b * rowsPad + i) * Npad;
  if (i >= N) {
    for (int j = lane; j < Npad; j += WAVE) arow[j] = (_Float16)0.f;
    if (lane == 0) deg[(size_t)b * rowsPad + i] = 1.f;
    return;
  }
  const float* srow = adj + ((size_t)b * N + i) * N;
  float m = -INFINITY; int cnt = 0;
  for (int j = lane; j < N; j += WAVE) {
    float v = srow[j];
    if (j == i) v = (v == 0.f) ? 1.f : v;       // add_remaining_self_loops(fill=1)
    if (v != 0.f) { cnt++; if (v > m) m = v; }
  }
  for (int s = 16; s; s >>= 1) {
    float o = __shfl_xor(m, s, WAVE); if (o > m) m = o;
    cnt += __shfl_xor(cnt, s, WAVE);
  }
  float sum = 0.f;
  for (int j = lane; j < N; j += WAVE) {
    float v = srow[j];
    if (j == i) v = (v == 0.f) ? 1.f : v;
    sum += (v != 0.f) ? __expf(v - m) : 0.f;
  }
  for (int s = 16; s; s >>= 1) sum += __shfl_xor(sum, s, WAVE);
  float inv = 1.f / sum;
  for (int j = lane; j < Npad; j += WAVE) {
    float a = 0.f;
    if (j < N) {
      float v = srow[j];
      if (j == i) v = (v == 0.f) ? 1.f : v;
      a = (v != 0.f) ? __expf(v - m) * inv : 0.f;
    }
    arow[j] = (_Float16)a;
  }
  if (lane == 0) deg[(size_t)b * rowsPad + i] = (float)cnt;
}

// ---------------------------------------------------------------------------
// K2b: xc[b,i,o] = (alpha[b,i,:] @ xt[b,:,o]) / deg[b,i] + bias[o]
// One wave per (batch, 16-row tile). WMMA f32 = f16 x f16, K-chunks of 32.
// A/B tiles staged to LDS with coalesced float4 copies, repacked per ISA
// 7.12.2 layouts (A: lane = M + 16*s, half h -> K=(h<8?h:h+8)+8*s;
//                 B: lane = N + 16*s, half h -> same K as A; C/D: vgpr=M%8, lane = N + 16*(M>=8)).
// ---------------------------------------------------------------------------
__global__ void __launch_bounds__(32)
k_aggregate_wmma(const _Float16* __restrict__ alphaH, // [B,rowsPad,Npad]
                 const _Float16* __restrict__ xtH,    // [B,Npad,32]
                 const float*    __restrict__ deg,    // [B,rowsPad]
                 const float*    __restrict__ bias,   // [32]
                 float*          __restrict__ xc,     // [B,N,32]
                 int N, int Npad, int rowsPad, int nTiles) {
  __shared__ _Float16 lA[16 * 32];
  __shared__ _Float16 lB[32 * 32];
  int b    = blockIdx.x / nTiles;
  int tile = blockIdx.x % nTiles;
  int lane = threadIdx.x;
  const _Float16* Abase = alphaH + ((size_t)b * rowsPad + tile * 16) * Npad;
  const _Float16* Bbase = xtH + (size_t)b * Npad * 32;
  v8f c0 = {}; v8f c1 = {};
  int nK = Npad / 32;
  for (int kc = 0; kc < nK; ++kc) {
    __syncthreads();
    // stage A tile (16 rows x 32 halves = 64 float4)
    for (int q = 0; q < 2; ++q) {
      int idx = q * 32 + lane;
      int ho  = idx * 8;
      int m = ho >> 5, k = ho & 31;
      ((float4*)lA)[idx] = *(const float4*)(Abase + (size_t)m * Npad + kc * 32 + k);
    }
    // stage B tile (32 K-rows x 32 cols = 128 float4)
    for (int q = 0; q < 4; ++q) {
      int idx = q * 32 + lane;
      int ho  = idx * 8;
      int j = ho >> 5, o = ho & 31;
      ((float4*)lB)[idx] = *(const float4*)(Bbase + (size_t)(kc * 32 + j) * 32 + o);
    }
    __syncthreads();
    int s = lane >> 4, mrow = lane & 15;
    v16h a, b0, b1;
    for (int h = 0; h < 16; ++h) {
      int kl = (h < 8 ? h : h + 8) + 8 * s;
      a[h]  = lA[mrow * 32 + kl];
      b0[h] = lB[kl * 32 + mrow];        // cols o = 0..15
      b1[h] = lB[kl * 32 + 16 + mrow];   // cols o = 16..31
    }
    c0 = __builtin_amdgcn_wmma_f32_16x16x32_f16(false, a, false, b0, (short)0, c0, false, false);
    c1 = __builtin_amdgcn_wmma_f32_16x16x32_f16(false, a, false, b1, (short)0, c1, false, false);
  }
  int ncol = lane & 15, hi = (lane >> 4) ? 8 : 0;
  for (int r = 0; r < 8; ++r) {
    int i = tile * 16 + r + hi;
    if (i < N) {
      float dinv = 1.f / deg[(size_t)b * rowsPad + i];
      xc[((size_t)b * N + i) * 32 + ncol]      = c0[r] * dinv + bias[ncol];
      xc[((size_t)b * N + i) * 32 + 16 + ncol] = c1[r] * dinv + bias[16 + ncol];
    }
  }
}

// ---------------------------------------------------------------------------
// K3: top-k pool (exact jax.lax.top_k semantics via stable rank), sigmoid gate,
// x/adj/orig gather, plus [gmp,gap] readout into hbuf. One block per graph.
// ---------------------------------------------------------------------------
__global__ void k_pool(const float* __restrict__ xc,    // [B,N,32]
                       const float* __restrict__ adjIn, // [B,N,N]
                       const int*   __restrict__ origIn,// [B,N] or null
                       const float* __restrict__ pw,    // [32]
                       float* __restrict__ xp,          // [B,k,32]
                       float* __restrict__ adjOut,      // [B,k,k] or null
                       int*   __restrict__ origOut,     // [B,k]
                       float* __restrict__ hbuf,        // [B,192]
                       int N, int k, int hoff) {
  __shared__ float sc[116];
  __shared__ int   sel[58];
  __shared__ float xg[58 * 32];
  int b = blockIdx.x, tid = threadIdx.x;
  float wn = 0.f;
  for (int o = 0; o < 32; ++o) wn += pw[o] * pw[o];
  wn = sqrtf(wn);
  for (int n = tid; n < N; n += blockDim.x) {
    float d = 0.f;
    const float* row = xc + ((size_t)b * N + n) * 32;
    for (int o = 0; o < 32; ++o) d += row[o] * pw[o];
    sc[n] = d / wn;
  }
  __syncthreads();
  for (int n = tid; n < N; n += blockDim.x) {
    float sn = sc[n];
    int r = 0;
    for (int j = 0; j < N; ++j) {
      float sj = sc[j];
      r += (sj > sn) || (sj == sn && j < n);
    }
    if (r < k) {
      sel[r] = n;
      float g = 1.f / (1.f + __expf(-sn));
      const float* row = xc + ((size_t)b * N + n) * 32;
      float* dst = xp + ((size_t)b * k + r) * 32;
      for (int o = 0; o < 32; ++o) {
        float v = row[o] * g;
        xg[r * 32 + o] = v;
        dst[o] = v;
      }
      origOut[b * k + r] = origIn ? origIn[b * N + n] : n;
    }
  }
  __syncthreads();
  if (adjOut) {
    for (int t = tid; t < k * k; t += blockDim.x) {
      int m1 = t / k, m2 = t % k;
      adjOut[(size_t)b * k * k + t] = adjIn[((size_t)b * N + sel[m1]) * N + sel[m2]];
    }
  }
  if (tid < 32) {
    float mx = -INFINITY, sm = 0.f;
    for (int m = 0; m < k; ++m) { float v = xg[m * 32 + tid]; mx = fmaxf(mx, v); sm += v; }
    hbuf[(size_t)b * 192 + hoff + tid]      = mx;
    hbuf[(size_t)b * 192 + hoff + 32 + tid] = sm / (float)k;
  }
}

// ---------------------------------------------------------------------------
// K4: augment: a = adjp + I; out = (a@a) with zeroed diagonal. One block/graph.
// ---------------------------------------------------------------------------
__global__ void k_augment(const float* __restrict__ adjp,
                          float* __restrict__ adja, int N) {
  __shared__ float a[58 * 58];
  int b = blockIdx.x, tid = threadIdx.x;
  for (int t = tid; t < N * N; t += blockDim.x) {
    float v = adjp[(size_t)b * N * N + t];
    if (t / N == t % N) v += 1.f;
    a[t] = v;
  }
  __syncthreads();
  for (int t = tid; t < N * N; t += blockDim.x) {
    int i = t / N, j = t % N;
    float acc = 0.f;
    if (i != j) {
      for (int u = 0; u < N; ++u) acc += a[i * N + u] * a[u * N + j];
    }
    adja[(size_t)b * N * N + t] = acc;
  }
}

// ---------------------------------------------------------------------------
// K5: MLP head: relu(h@fc1+b) -> eval BN -> @fc2+b.  One wave per graph.
// ---------------------------------------------------------------------------
__global__ void k_head(const float* __restrict__ hbuf, const float* __restrict__ fc1w,
                       const float* __restrict__ fc1b, const float* __restrict__ bng,
                       const float* __restrict__ bnb,  const float* __restrict__ bnm,
                       const float* __restrict__ bnv,  const float* __restrict__ fc2w,
                       const float* __restrict__ fc2b, float* __restrict__ out, int B) {
  int w    = blockIdx.x * (blockDim.x / WAVE) + (threadIdx.x / WAVE);
  int lane = threadIdx.x & (WAVE - 1);
  if (w >= B) return;
  const float* hr = hbuf + (size_t)w * 192;
  float t = 0.f;
  if (lane < 24) {
    for (int i = 0; i < 192; ++i) t += hr[i] * fc1w[i * 24 + lane];
    t += fc1b[lane];
    t = t > 0.f ? t : 0.f;
    t = (t - bnm[lane]) * rsqrtf(bnv[lane] + 1e-5f) * bng[lane] + bnb[lane];
  }
  for (int c = 0; c < 2; ++c) {
    float p = (lane < 24) ? t * fc2w[lane * 2 + c] : 0.f;
    for (int s = 16; s; s >>= 1) p += __shfl_xor(p, s, WAVE);
    if (lane == 0) out[(size_t)w * 2 + c] = p + fc2b[c];
  }
}

// ---------------------------------------------------------------------------
extern "C" void kernel_launch(void* const* d_in, const int* in_sizes, int n_in,
                              void* d_out, int out_size, void* d_ws, size_t ws_size,
                              hipStream_t stream) {
  const int B = 1024;
  const float* x    = (const float*)d_in[0];
  const float* adj  = (const float*)d_in[1];
  // d_in[2] (pos) is always one-hot identity -> handled analytically.
  const float* n1w1 = (const float*)d_in[3];
  const float* n1w2 = (const float*)d_in[4];
  const float* b1   = (const float*)d_in[5];
  const float* p1w  = (const float*)d_in[6];
  const float* n2w1 = (const float*)d_in[7];
  const float* n2w2 = (const float*)d_in[8];
  const float* b2   = (const float*)d_in[9];
  const float* p2w  = (const float*)d_in[10];
  const float* n3w1 = (const float*)d_in[11];
  const float* n3w2 = (const float*)d_in[12];
  const float* b3   = (const float*)d_in[13];
  const float* p3w  = (const float*)d_in[14];
  const float* fc1w = (const float*)d_in[15];
  const float* fc1b = (const float*)d_in[16];
  const float* bng  = (const float*)d_in[17];
  const float* bnb  = (const float*)d_in[18];
  const float* bnm  = (const float*)d_in[19];
  const float* bnv  = (const float*)d_in[20];
  const float* fc2w = (const float*)d_in[21];
  const float* fc2b = (const float*)d_in[22];
  float* out = (float*)d_out;

  // workspace carve (256B aligned)
  char* ws = (char*)d_ws;
  size_t off = 0;
  auto carve = [&](size_t bytes) -> char* {
    char* p = ws + off;
    off += (bytes + 255) & ~(size_t)255;
    return p;
  };
  float*    wall1  = (float*)   carve((size_t)116 * 116 * 32 * 4);
  float*    wall2  = (float*)   carve((size_t)116 * 32 * 32 * 4);
  float*    wall3  = (float*)   carve((size_t)116 * 32 * 32 * 4);
  _Float16* xtH    = (_Float16*)carve((size_t)B * 128 * 32 * 2);
  _Float16* alphaH = (_Float16*)carve((size_t)B * 128 * 128 * 2);
  float*    deg    = (float*)   carve((size_t)B * 128 * 4);
  float*    xc     = (float*)   carve((size_t)B * 116 * 32 * 4);
  float*    xp     = (float*)   carve((size_t)B * 58 * 32 * 4);
  float*    adjp   = (float*)   carve((size_t)B * 58 * 58 * 4);
  float*    adja   = (float*)   carve((size_t)B * 58 * 58 * 4);
  int*      origA  = (int*)     carve((size_t)B * 58 * 4);
  int*      origB  = (int*)     carve((size_t)B * 29 * 4);
  float*    hbuf   = (float*)   carve((size_t)B * 192 * 4);
  (void)ws_size; (void)in_sizes; (void)n_in; (void)out_size;

  // precompute per-original-node dynamic weights for all 3 conv layers
  k_precompute_wall<<<116, 256, 0, stream>>>(n1w1, n1w2, wall1, 116);
  k_precompute_wall<<<116, 256, 0, stream>>>(n2w1, n2w2, wall2, 32);
  k_precompute_wall<<<116, 256, 0, stream>>>(n3w1, n3w2, wall3, 32);

  // ---- Layer 1: N=116, Npad=128, rowsPad=128, nTiles=8, Ci=116
  k_softmax       <<<B * 128 / 4, 128, 0, stream>>>(adj, alphaH, deg, 116, 128, 128);
  k_conv_xt       <<<B * 128 / 4, 128, 0, stream>>>(x, nullptr, wall1, xtH, 116, 128, 116);
  k_aggregate_wmma<<<B * 8, 32, 0, stream>>>(alphaH, xtH, deg, b1, xc, 116, 128, 128, 8);
  k_pool          <<<B, 128, 0, stream>>>(xc, adj, nullptr, p1w, xp, adjp, origA, hbuf, 116, 58, 0);

  // ---- Layer 2: N=58, Npad=64, rowsPad=64, nTiles=4, Ci=32
  k_augment       <<<B, 128, 0, stream>>>(adjp, adja, 58);
  k_softmax       <<<B * 64 / 4, 128, 0, stream>>>(adja, alphaH, deg, 58, 64, 64);
  k_conv_xt       <<<B * 64 / 4, 128, 0, stream>>>(xp, origA, wall2, xtH, 58, 64, 32);
  k_aggregate_wmma<<<B * 4, 32, 0, stream>>>(alphaH, xtH, deg, b2, xc, 58, 64, 64, 4);
  k_pool          <<<B, 128, 0, stream>>>(xc, adja, origA, p2w, xp, adjp, origB, hbuf, 58, 29, 64);

  // ---- Layer 3: N=29, Npad=32, rowsPad=32, nTiles=2, Ci=32
  k_augment       <<<B, 128, 0, stream>>>(adjp, adja, 29);
  k_softmax       <<<B * 32 / 4, 128, 0, stream>>>(adja, alphaH, deg, 29, 32, 32);
  k_conv_xt       <<<B * 32 / 4, 128, 0, stream>>>(xp, origB, wall3, xtH, 29, 32, 32);
  k_aggregate_wmma<<<B * 2, 32, 0, stream>>>(alphaH, xtH, deg, b3, xc, 29, 32, 32, 2);
  // final pool: adj/orig outputs unused downstream -> adjOut null, orig scratch
  k_pool          <<<B, 128, 0, stream>>>(xc, adja, origB, p3w, xp, nullptr, origA, hbuf, 29, 15, 128);

  // ---- head
  k_head<<<B / 4, 128, 0, stream>>>(hbuf, fc1w, fc1b, bng, bnb, bnm, bnv, fc2w, fc2b, out, B);
}
